// MaskedAttentionHead_3229815407300
// MI455X (gfx1250) — matrix operand order repaired and checked by reading, not compiled, not run on hardware
//
#include <hip/hip_runtime.h>

// ---- problem constants (MaskedAttentionHead) ----
#define D_MODEL 1024
#define DK      64
#define S_LEN   4096
#define BATCH   4

typedef __attribute__((ext_vector_type(16))) __bf16 v16bf;
typedef __attribute__((ext_vector_type(8)))  __bf16 v8bf;
typedef __attribute__((ext_vector_type(4)))  __bf16 v4bf;
typedef __attribute__((ext_vector_type(2)))  __bf16 v2bf;
typedef __attribute__((ext_vector_type(8)))  float  v8f;
typedef __attribute__((ext_vector_type(4)))  float  v4f;
typedef __attribute__((ext_vector_type(4)))  unsigned int u32x4;
typedef __attribute__((ext_vector_type(8)))  int    i32x8;
typedef __attribute__((ext_vector_type(4)))  int    i32x4;

// ---- float -> bf16 (RNE) ----------------------------------------------------
// Prefer the packed-cvt builtin; otherwise rely on clang's fptrunc->bfloat
// lowering (backend picks the native cvt when the target has one).
#if __has_builtin(__builtin_amdgcn_cvt_pk_bf16_f32)
__device__ __forceinline__ __bf16 f2bf(float x) {
    v2bf t = __builtin_amdgcn_cvt_pk_bf16_f32(x, 0.0f);
    return t[0];
}
__device__ __forceinline__ v2bf f2bf2(float x, float y) {
    return __builtin_amdgcn_cvt_pk_bf16_f32(x, y);
}
#else
__device__ __forceinline__ __bf16 f2bf(float x) { return (__bf16)x; }
__device__ __forceinline__ v2bf f2bf2(float x, float y) {
    v2bf r; r[0] = (__bf16)x; r[1] = (__bf16)y; return r;
}
#endif

__device__ __forceinline__ v16bf mk16(v8bf lo, v8bf hi) {
    v16bf r;
#pragma unroll
    for (int i = 0; i < 8; ++i) { r[i] = lo[i]; r[i + 8] = hi[i]; }
    return r;
}

// ---- Tensor Data Mover: 2-D tile (bf16) global -> LDS ------------------------
// D# layout per CDNA5 ISA (2-D: groups 2/3 zero).
#if __has_builtin(__builtin_amdgcn_tensor_load_to_lds)
#define HAVE_TDM 1
#if __has_include(<hip/amd_detail/amd_gfx1250_TDM.h>)
#define TDM_CALL(g0, g1, z4) \
    __builtin_amdgcn_tensor_load_to_lds(g0, g1, z4, z4, (i32x8){0,0,0,0,0,0,0,0}, 0)
#else
#define TDM_CALL(g0, g1, z4) \
    __builtin_amdgcn_tensor_load_to_lds(g0, g1, z4, z4, 0)
#endif
__device__ __forceinline__ void tdm_load_tile_bf16(const void* lds_dst, const void* gsrc,
                                                   int tile_d0, int tile_d1,
                                                   int ten_d0, int ten_d1, int stride0) {
    // generic LDS addr[31:0] == LDS byte offset (aperture mapping, ISA §10.2)
    unsigned lds = (unsigned)(unsigned long long)(__UINTPTR_TYPE__)lds_dst;
    unsigned long long ga = (unsigned long long)(__UINTPTR_TYPE__)gsrc;
    u32x4 g0;
    g0[0] = 1u;                                   // count=1, user descriptor
    g0[1] = lds;                                  // lds_addr
    g0[2] = (unsigned)ga;                         // global_addr[31:0]
    g0[3] = (unsigned)(ga >> 32) | (2u << 30);    // global_addr[56:32] | type=2
    i32x8 g1;
    g1[0] = 1 << 16;                              // data_size=1 -> 2 bytes/elem
    g1[1] = (ten_d0 & 0xFFFF) << 16;              // tensor_dim0[15:0]
    g1[2] = ((unsigned)ten_d0 >> 16) | ((ten_d1 & 0xFFFF) << 16);
    g1[3] = ((unsigned)ten_d1 >> 16) | (tile_d0 << 16);   // tile_dim0
    g1[4] = tile_d1 & 0xFFFF;                     // tile_dim1 (tile_dim2=0)
    g1[5] = stride0;                              // tensor_dim0_stride[31:0]
    g1[6] = 0;
    g1[7] = 0;
    i32x4 z4 = {0, 0, 0, 0};
    TDM_CALL(g0, g1, z4);
}
#else
#define HAVE_TDM 0
#endif

// ============================================================================
// Kernel 1: X[M=16384,1024] @ W[1024,64] + b -> bf16 out[M,64]
// 256 threads (8 waves). Block tile: 128 rows x 64 cols. Wave tile: 16 x 64.
// (HBM-roofline phase: 192 MB of f32 read once, emitted as bf16.)
// ============================================================================
__global__ __launch_bounds__(256)
void proj_bf16_kernel(const float* __restrict__ X, const float* __restrict__ W,
                      const float* __restrict__ bias, __bf16* __restrict__ Out)
{
    __shared__ alignas(16) __bf16 Xs[128 * 32];  // [row][k]   8 KB
    __shared__ alignas(16) __bf16 Wt[64 * 32];   // [n][k]     4 KB (transposed)

    const int tid  = threadIdx.x;
    const int lane = tid & 31;
    const int wid  = tid >> 5;        // 0..7
    const int hgrp = lane >> 4;       // 0/1: half-group within wave
    const int l15  = lane & 15;
    const int row0 = blockIdx.x * 128;

    v8f acc[4];
#pragma unroll
    for (int t = 0; t < 4; ++t)
#pragma unroll
        for (int i = 0; i < 8; ++i) acc[t][i] = 0.0f;

    for (int k0 = 0; k0 < D_MODEL; k0 += 32) {
        // --- stage X panel: 128x32 f32 -> bf16 (16 elems/thread, 4x float4) ---
#pragma unroll
        for (int c = 0; c < 4; ++c) {
            int flat = tid * 16 + c * 4;          // 0..4095
            int row = flat >> 5, kk = flat & 31;
            v4f xv = *(const v4f*)(X + (size_t)(row0 + row) * D_MODEL + k0 + kk);
            v2bf lo = f2bf2(xv[0], xv[1]);
            v2bf hi = f2bf2(xv[2], xv[3]);
            v4bf xb;
            xb[0] = lo[0]; xb[1] = lo[1]; xb[2] = hi[0]; xb[3] = hi[1];
            *(v4bf*)(Xs + row * 32 + kk) = xb;
        }
        // --- stage W panel transposed: [32k][64n] f32 -> Wt[n][k] bf16 ---
#pragma unroll
        for (int c = 0; c < 2; ++c) {
            int flat = tid * 8 + c * 4;           // 0..2047
            int kk = flat >> 6, nn = flat & 63;
            v4f wv = *(const v4f*)(W + (size_t)(k0 + kk) * DK + nn);
#pragma unroll
            for (int i = 0; i < 4; ++i) Wt[(nn + i) * 32 + kk] = f2bf(wv[i]);
        }
        __syncthreads();

        // A fragment: rows [wid*16, wid*16+16), K = {b..b+7, b+16..b+23}, b=8*hgrp
        const int arow = wid * 16 + l15;
        const int kb = hgrp * 8;
        v8bf a0 = *(const v8bf*)(Xs + arow * 32 + kb);
        v8bf a1 = *(const v8bf*)(Xs + arow * 32 + kb + 16);
        v16bf afrag = mk16(a0, a1);

#pragma unroll
        for (int nt = 0; nt < 4; ++nt) {
            const int n = nt * 16 + l15;
            v8bf b0 = *(const v8bf*)(Wt + n * 32 + hgrp * 8);
            v8bf b1 = *(const v8bf*)(Wt + n * 32 + hgrp * 8 + 16);
            v16bf bfrag = mk16(b0, b1);
            acc[nt] = __builtin_amdgcn_wmma_f32_16x16x32_bf16(
                false, afrag, false, bfrag, (short)0, acc[nt], false, false);
        }
        __syncthreads();
    }

#pragma unroll
    for (int nt = 0; nt < 4; ++nt) {
        const int col = nt * 16 + l15;
        const float bv = bias[col];
#pragma unroll
        for (int r = 0; r < 8; ++r) {
            const int row = row0 + wid * 16 + hgrp * 8 + r;
            Out[(size_t)row * DK + col] = f2bf(acc[nt][r] + bv);
        }
    }
}

// ============================================================================
// Kernel 2: flash attention, pre-softmax multiplicative row mask.
// 128 threads (4 waves). Block: 64 queries. Wave: 16 queries. d_k = 64.
// K tiles: TDM (tensor_load_to_lds), double-buffered, TENSORcnt-gated.
// V tiles: cooperative transposed staging (B-frags stay ds_load_b128).
// ============================================================================
__global__ __launch_bounds__(128)
void flash_attn_kernel(const __bf16* __restrict__ Qh, const __bf16* __restrict__ Kh,
                       const __bf16* __restrict__ Vh, const float* __restrict__ Mk,
                       float* __restrict__ Y)
{
    __shared__ alignas(16) __bf16 Ks[2][64 * 64];   // [buf][key][dk] 16 KB
    __shared__ alignas(16) __bf16 Vt[64 * 64];      // [dk][key]       8 KB
    __shared__ alignas(16) __bf16 Ps[4][16 * 64];   // per-wave P      8 KB

    const int tid  = threadIdx.x;
    const int lane = tid & 31;
    const int wid  = tid >> 5;       // 0..3
    const int hgrp = lane >> 4;
    const int l15  = lane & 15;
    const int bb   = blockIdx.y;
    const int qw   = blockIdx.x * 64 + wid * 16;
    const size_t bS = (size_t)bb * S_LEN;

    // --- Q fragments (A-layout), straight from global bf16 qh ---
    v16bf qf[2];
    {
        const __bf16* qrow = Qh + (bS + qw + l15) * DK;
#pragma unroll
        for (int f = 0; f < 2; ++f) {
            int base = f * 32 + hgrp * 8;
            v8bf lo = *(const v8bf*)(qrow + base);
            v8bf hi = *(const v8bf*)(qrow + base + 16);
            qf[f] = mk16(lo, hi);
        }
    }

    const float scale = 0.125f;           // 1/sqrt(64)
    float cm[8];
#pragma unroll
    for (int r = 0; r < 8; ++r) cm[r] = scale * Mk[bS + qw + hgrp * 8 + r];

    float mrow[8], lrow[8];
#pragma unroll
    for (int r = 0; r < 8; ++r) { mrow[r] = -__builtin_inff(); lrow[r] = 0.0f; }
    v8f O[4];
#pragma unroll
    for (int t = 0; t < 4; ++t)
#pragma unroll
        for (int i = 0; i < 8; ++i) O[t][i] = 0.0f;

#if HAVE_TDM
    // prologue: K tile 0 -> buffer 0 via TDM (one wave issues for the WG)
    if (wid == 0) {
        tdm_load_tile_bf16(&Ks[0][0], Kh + bS * DK, DK, 64, DK, BATCH * S_LEN, DK);
    }
#endif

    for (int kt = 0; kt < S_LEN / 64; ++kt) {
        const int key0 = kt * 64;
        const int cur  = kt & 1;

#if !HAVE_TDM
        // fallback: cooperative K staging
#pragma unroll
        for (int c = 0; c < 4; ++c) {
            int flat = (tid * 4 + c) * 8;
            int key = flat >> 6, dk = flat & 63;
            v8bf kv = *(const v8bf*)(Kh + (bS + key0 + key) * DK + dk);
            *(v8bf*)(&Ks[cur][0] + key * 64 + dk) = kv;
        }
#endif
        // --- stage V transposed: Vt[dk][key] (feeds contiguous B-frag loads) ---
#pragma unroll
        for (int c = 0; c < 4; ++c) {
            int flat = (tid * 4 + c) * 8;        // 0..4095, step 8
            int key = flat >> 6, dk = flat & 63;
            v8bf vv = *(const v8bf*)(Vh + (bS + key0 + key) * DK + dk);
#pragma unroll
            for (int i = 0; i < 8; ++i) Vt[(dk + i) * 64 + key] = vv[i];
        }
        if (kt + 1 < S_LEN / 64) {  // pull next V tile toward L0/L2
            __builtin_prefetch(Vh + (bS + key0 + 64 + (tid >> 1)) * DK + (tid & 1) * 32, 0, 1);
        }

#if HAVE_TDM
        if (wid == 0) __builtin_amdgcn_s_wait_tensorcnt(0);  // K tile kt landed
#endif
        __syncthreads();                                     // (a) publish Ks/Vt

#if HAVE_TDM
        if (kt + 1 < S_LEN / 64 && wid == 0) {
            // next K tile into the other buffer; its old readers passed (a)
            tdm_load_tile_bf16(&Ks[cur ^ 1][0], Kh + (bS + key0 + 64) * DK,
                               DK, 64, DK, BATCH * S_LEN, DK);
        }
#endif

        // --- S = Q K^T : 16x64 scores per wave (4 col-tiles x 2 k-steps) ---
        v8f sacc[4];
#pragma unroll
        for (int t = 0; t < 4; ++t)
#pragma unroll
            for (int i = 0; i < 8; ++i) sacc[t][i] = 0.0f;
#pragma unroll
        for (int nt = 0; nt < 4; ++nt) {
            const int key = nt * 16 + l15;
#pragma unroll
            for (int ks = 0; ks < 2; ++ks) {
                const __bf16* kr = &Ks[cur][0] + key * 64 + ks * 32 + hgrp * 8;
                v16bf bfrag = mk16(*(const v8bf*)kr, *(const v8bf*)(kr + 16));
                sacc[nt] = __builtin_amdgcn_wmma_f32_16x16x32_bf16(
                    false, qf[ks], false, bfrag, (short)0, sacc[nt], false, false);
            }
        }

        // --- online softmax (row = hgrp*8 + r; 16-lane half-group reductions) ---
        float rmax[8];
#pragma unroll
        for (int r = 0; r < 8; ++r) {
            float v = sacc[0][r] * cm[r];
#pragma unroll
            for (int t = 1; t < 4; ++t) v = fmaxf(v, sacc[t][r] * cm[r]);
#pragma unroll
            for (int xm = 1; xm < 16; xm <<= 1) v = fmaxf(v, __shfl_xor(v, xm, 32));
            rmax[r] = v;
        }
        float alpha[8], rsum[8];
#pragma unroll
        for (int r = 0; r < 8; ++r) {
            float mn = fmaxf(mrow[r], rmax[r]);
            alpha[r] = __expf(mrow[r] - mn);
            mrow[r] = mn;
            rsum[r] = 0.0f;
        }
#pragma unroll
        for (int t = 0; t < 4; ++t) {
            const int col = t * 16 + l15;
#pragma unroll
            for (int r = 0; r < 8; ++r) {
                float p = __expf(sacc[t][r] * cm[r] - mrow[r]);
                rsum[r] += p;
                Ps[wid][(hgrp * 8 + r) * 64 + col] = f2bf(p);
            }
        }
#pragma unroll
        for (int r = 0; r < 8; ++r) {
            float v = rsum[r];
#pragma unroll
            for (int xm = 1; xm < 16; xm <<= 1) v += __shfl_xor(v, xm, 32);
            lrow[r] = lrow[r] * alpha[r] + v;
        }
#pragma unroll
        for (int t = 0; t < 4; ++t)
#pragma unroll
            for (int r = 0; r < 8; ++r) O[t][r] *= alpha[r];

        // --- P fragments (per-wave LDS buffer is in-order; no barrier) ---
        v16bf pf[2];
        const __bf16* prow = Ps[wid] + l15 * 64;
#pragma unroll
        for (int f = 0; f < 2; ++f) {
            int base = f * 32 + hgrp * 8;
            pf[f] = mk16(*(const v8bf*)(prow + base), *(const v8bf*)(prow + base + 16));
        }

        // --- O += P V : B-frags are contiguous rows of Vt[dk][key] ---
#pragma unroll
        for (int nt = 0; nt < 4; ++nt) {
            const int dk = nt * 16 + l15;
#pragma unroll
            for (int ks = 0; ks < 2; ++ks) {
                const __bf16* vr = Vt + dk * 64 + ks * 32 + hgrp * 8;
                v16bf bfrag = mk16(*(const v8bf*)vr, *(const v8bf*)(vr + 16));
                O[nt] = __builtin_amdgcn_wmma_f32_16x16x32_bf16(
                    false, pf[ks], false, bfrag, (short)0, O[nt], false, false);
            }
        }
        __syncthreads();   // (b) protect Vt (and K fallback buffer) for restage
    }

    // --- y = O / l, f32 out ---
#pragma unroll
    for (int nt = 0; nt < 4; ++nt) {
        const int col = nt * 16 + l15;
#pragma unroll
        for (int r = 0; r < 8; ++r) {
            const int row = qw + hgrp * 8 + r;
            Y[(bS + row) * DK + col] = O[nt][r] / lrow[r];
        }
    }
}

// ============================================================================
extern "C" void kernel_launch(void* const* d_in, const int* in_sizes, int n_in,
                              void* d_out, int out_size, void* d_ws, size_t ws_size,
                              hipStream_t stream)
{
    const float* q  = (const float*)d_in[0];
    const float* k  = (const float*)d_in[1];
    const float* v  = (const float*)d_in[2];
    const float* m  = (const float*)d_in[3];
    const float* Wq = (const float*)d_in[4];
    const float* bq = (const float*)d_in[5];
    const float* Wk = (const float*)d_in[6];
    const float* bk = (const float*)d_in[7];
    const float* Wv = (const float*)d_in[8];
    const float* bv = (const float*)d_in[9];

    const size_t headElems = (size_t)BATCH * S_LEN * DK;
    __bf16* qh = (__bf16*)d_ws;
    __bf16* kh = qh + headElems;
    __bf16* vh = kh + headElems;

    dim3 pgrid(BATCH * S_LEN / 128);   // 128 blocks
    proj_bf16_kernel<<<pgrid, 256, 0, stream>>>(q, Wq, bq, qh);
    proj_bf16_kernel<<<pgrid, 256, 0, stream>>>(k, Wk, bk, kh);
    proj_bf16_kernel<<<pgrid, 256, 0, stream>>>(v, Wv, bv, vh);

    dim3 agrid(S_LEN / 64, BATCH);     // 256 blocks
    flash_attn_kernel<<<agrid, 128, 0, stream>>>(qh, kh, vh, m, (float*)d_out);
}
